// HungarianMatcher_18313740550407
// MI455X (gfx1250) — compile-verified, matrix operand order checked by reading, not run on hardware
//
#include <hip/hip_runtime.h>

typedef __attribute__((ext_vector_type(16))) _Float16 v16h;
typedef __attribute__((ext_vector_type(8)))  float    v8f;

#define B_   128
#define Q_   128
#define G_   24
#define GPAD 32   // padded G stride for the cost workspace

// ---------------------------------------------------------------------------
// Kernel 0: zero the cost accumulation workspace (it is atomically added into)
// ---------------------------------------------------------------------------
__global__ void zero_kernel(float* __restrict__ p, int n) {
  int i = blockIdx.x * blockDim.x + threadIdx.x;
  if (i < n) p[i] = 0.0f;
}

// ---------------------------------------------------------------------------
// Kernel 1: per (batch, 16-row q-tile, tensor) softmax-gather.
//   - stream the 16xV tile once (row max in registers)
//   - exp(x - max) -> LDS (f16)
//   - denominator = rowsum via v_wmma_f32_16x16x32_f16 with an all-ones B
//     (ones-B makes the WMMA K layout irrelevant; every output column is the
//      exact row sum; lanes 0 and 16 hold rows 0-7 / 8-15 in their C regs).
//     A-fragments are double-buffered so the next ds_load_b128 pair is in
//     flight while the current WMMA executes (wait dscnt<=2, not 0).
//   - gather 24 gold entries from (cached) global, atomicAdd(-p) into cost
// ---------------------------------------------------------------------------
template <int V>
__device__ void cost_tensor_tile(int lane, int b, int q0,
                                 const float* __restrict__ logits,
                                 const int*   __restrict__ gold,
                                 float* __restrict__ cost,
                                 _Float16* expTile, float* rowMax, float* rowSum)
{
  constexpr int VL = V / 32;
  constexpr int KSTEPS = V / 32;

  // ---- pass 1: row max + exp -> LDS -------------------------------------
  for (int r = 0; r < 16; ++r) {
    const float* row = logits + ((size_t)(b * Q_ + q0 + r)) * V;
    if (r + 1 < 16) __builtin_prefetch(row + V, 0, 1);   // global_prefetch_b8
    float x[VL];
    float mx = -3.0e38f;
#pragma unroll
    for (int i = 0; i < VL; ++i) {
      float vv = row[lane + 32 * i];
      vv = (vv != vv) ? -1.0e9f : vv;      // NaN -> -1e9 (matches reference)
      x[i] = vv;
      mx = fmaxf(mx, vv);
    }
#pragma unroll
    for (int off = 16; off > 0; off >>= 1)
      mx = fmaxf(mx, __shfl_xor(mx, off, 32));
#pragma unroll
    for (int i = 0; i < VL; ++i)
      expTile[r * V + lane + 32 * i] = (_Float16)__expf(x[i] - mx);
    if (lane == 0) rowMax[r] = mx;
  }
  __syncthreads();

  // ---- denominator: WMMA rowsum against all-ones B ----------------------
  v16h bones;
#pragma unroll
  for (int i = 0; i < 16; ++i) bones[i] = (_Float16)1.0f;

  v8f c = {};
  const int m = lane & 15;   // A-matrix row owned by this lane
  const int h = lane >> 4;   // which K-half this lane supplies
  const _Float16* arow = &expTile[m * V + 16 * h];

  v16h a0 = *(const v16h*)&arow[0];                 // prologue fragment
#pragma unroll
  for (int k = 0; k < KSTEPS; ++k) {
    v16h a1;
    if (k + 1 < KSTEPS) a1 = *(const v16h*)&arow[(k + 1) * 32];  // prefetch
    c = __builtin_amdgcn_wmma_f32_16x16x32_f16(false, a0, false, bones,
                                               (short)0, c, false, false);
    a0 = a1;
  }
  if ((lane & 15) == 0) {
#pragma unroll
    for (int r = 0; r < 8; ++r) rowSum[r + 8 * h] = c[r];
  }
  __syncthreads();

  // ---- gather 24 gold probabilities, accumulate -p ----------------------
  if (lane < G_) {
    const int idx = gold[b * G_ + lane];
    for (int r = 0; r < 16; ++r) {
      float vv = logits[((size_t)(b * Q_ + q0 + r)) * V + idx];
      vv = (vv != vv) ? -1.0e9f : vv;
      float p = __expf(vv - rowMax[r]) / rowSum[r];
      atomicAdd(&cost[((size_t)(b * Q_ + q0 + r)) * GPAD + lane], -p);
    }
  }
}

__global__ __launch_bounds__(32) void cost_kernel(
    const float* __restrict__ pr, const float* __restrict__ hs,
    const float* __restrict__ he, const float* __restrict__ ts,
    const float* __restrict__ te,
    const int* __restrict__ g_rel, const int* __restrict__ g_hs,
    const int* __restrict__ g_he,  const int* __restrict__ g_ts,
    const int* __restrict__ g_te,
    float* __restrict__ cost)
{
  __shared__ alignas(32) _Float16 expTile[16 * 1024];
  __shared__ float rowMax[16];
  __shared__ float rowSum[16];

  const int lane = threadIdx.x;
  const int t  = blockIdx.x % 5;
  const int qt = blockIdx.x / 5;
  const int b  = blockIdx.y;
  const int q0 = qt * 16;

  switch (t) {
    case 0: cost_tensor_tile<256 >(lane, b, q0, pr, g_rel, cost, expTile, rowMax, rowSum); break;
    case 1: cost_tensor_tile<1024>(lane, b, q0, hs, g_hs,  cost, expTile, rowMax, rowSum); break;
    case 2: cost_tensor_tile<1024>(lane, b, q0, he, g_he,  cost, expTile, rowMax, rowSum); break;
    case 3: cost_tensor_tile<1024>(lane, b, q0, ts, g_ts,  cost, expTile, rowMax, rowSum); break;
    default: cost_tensor_tile<1024>(lane, b, q0, te, g_te, cost, expTile, rowMax, rowSum); break;
  }
}

// ---------------------------------------------------------------------------
// Kernel 2: Jonker-Volgenant assignment (e-maxx formulation), one wave per
// batch. n = 24 rows (gold), m = 128 cols (pred). O(m) scans parallelized
// 4 columns/lane with shfl min+argmin reductions.
// ---------------------------------------------------------------------------
__global__ __launch_bounds__(32) void hungarian_kernel(
    const float* __restrict__ cost, int* __restrict__ out)
{
  __shared__ float a[25][129];     // 1-indexed padded cost, a[row][col]
  __shared__ float u[32];
  __shared__ float v[129];
  __shared__ float minv[129];
  __shared__ int   p[129];
  __shared__ int   way[129];
  __shared__ int   used[129];

  const int lane = threadIdx.x;
  const int b = blockIdx.x;
  const float INF = 1.0e9f;

  // load transposed cost: a[i][j] = cost[b][j-1][i-1]
  for (int e = lane; e < G_ * Q_; e += 32) {
    int i = e / Q_ + 1;
    int j = e % Q_ + 1;
    a[i][j] = cost[((size_t)(b * Q_) + (j - 1)) * GPAD + (i - 1)];
  }
  for (int j = lane; j <= Q_; j += 32) { v[j] = 0.0f; p[j] = 0; }
  if (lane < G_ + 1) u[lane] = 0.0f;
  __syncthreads();

  for (int i = 1; i <= G_; ++i) {
    if (lane == 0) p[0] = i;
    for (int j = lane; j <= Q_; j += 32) { minv[j] = INF; used[j] = 0; way[j] = 0; }
    __syncthreads();

    int j0 = 0;
    while (true) {
      const int i0 = p[j0];
      if (lane == 0) used[j0] = 1;
      __syncthreads();

      const float ui0 = u[i0];
      float bval = INF;
      int   bidx = 0;
      for (int j = 1 + lane; j <= Q_; j += 32) {
        if (!used[j]) {
          float cur = a[i0][j] - ui0 - v[j];
          if (cur < minv[j]) { minv[j] = cur; way[j] = j0; }
          float mv = minv[j];
          if (mv < bval) { bval = mv; bidx = j; }
        }
      }
#pragma unroll
      for (int off = 16; off > 0; off >>= 1) {
        float ov = __shfl_xor(bval, off, 32);
        int   oi = __shfl_xor(bidx, off, 32);
        if (ov < bval || (ov == bval && oi < bidx)) { bval = ov; bidx = oi; }
      }
      const float delta = bval;
      const int j1 = bidx;

      for (int j = lane; j <= Q_; j += 32) {
        if (used[j]) { u[p[j]] += delta; v[j] -= delta; }
        else         { minv[j] -= delta; }
      }
      __syncthreads();

      j0 = j1;
      if (p[j0] == 0) break;   // reached a free column
    }

    // augmenting-path reconstruction (wave-uniform bookkeeping; lane 0 writes)
    if (lane == 0) {
      int jj = j0;
      while (jj != 0) {
        int jn = way[jj];
        p[jj] = p[jn];
        jj = jn;
      }
    }
    __syncthreads();
  }

  // emit in ascending pred-index order == reference's argsort(col4row)
  if (lane == 0) {
    int k = 0;
    for (int j = 1; j <= Q_; ++j) {
      if (p[j] != 0) {
        out[b * G_ + k]             = j - 1;      // row_ind (pred index)
        out[B_ * G_ + b * G_ + k]   = p[j] - 1;   // col_ind (gold index)
        ++k;
      }
    }
  }
}

// ---------------------------------------------------------------------------
extern "C" void kernel_launch(void* const* d_in, const int* in_sizes, int n_in,
                              void* d_out, int out_size, void* d_ws, size_t ws_size,
                              hipStream_t stream) {
  const float* pr = (const float*)d_in[0];
  const float* hs = (const float*)d_in[1];
  const float* he = (const float*)d_in[2];
  const float* ts = (const float*)d_in[3];
  const float* te = (const float*)d_in[4];
  const int* g_rel = (const int*)d_in[5];
  const int* g_hs  = (const int*)d_in[6];
  const int* g_he  = (const int*)d_in[7];
  const int* g_ts  = (const int*)d_in[8];
  const int* g_te  = (const int*)d_in[9];

  float* cost = (float*)d_ws;                       // [B][Q][GPAD] f32 = 2 MB
  const int ncost = B_ * Q_ * GPAD;

  zero_kernel<<<(ncost + 255) / 256, 256, 0, stream>>>(cost, ncost);

  // 8 q-tiles x 5 tensors in x, batch in y; 1 wave per block
  cost_kernel<<<dim3(8 * 5, B_), 32, 0, stream>>>(
      pr, hs, he, ts, te, g_rel, g_hs, g_he, g_ts, g_te, cost);

  hungarian_kernel<<<B_, 32, 0, stream>>>(cost, (int*)d_out);
}
